// EncoderLayer_38087769981218
// MI455X (gfx1250) — matrix operand order repaired and checked by reading, not compiled, not run on hardware
//
#include <hip/hip_runtime.h>
#include <cstdint>

#define B_  4
#define S_  1024
#define D_  1024
#define H_  16
#define F_  4096
#define DK_ 64
#define M_  (B_ * S_)   // 4096 rows

typedef __attribute__((ext_vector_type(16))) __bf16 v16bf;
typedef __attribute__((ext_vector_type(8)))  __bf16 v8bf;
typedef __attribute__((ext_vector_type(4)))  __bf16 v4bf;
typedef __attribute__((ext_vector_type(8)))  float  v8f;

union Frag { v16bf v; v8bf h[2]; };

__device__ __forceinline__ __bf16 f2bf(float f) {
  unsigned u = __builtin_bit_cast(unsigned, f);
  unsigned r = u + 0x7FFFu + ((u >> 16) & 1u);   // round-to-nearest-even
  unsigned short hs = (unsigned short)(r >> 16);
  return __builtin_bit_cast(__bf16, hs);
}

__device__ __forceinline__ v8f wmma_bf16(const Frag& a, const Frag& b, v8f c) {
  // D = A(16x32 bf16) * B(32x16 bf16) + C(16x16 f32)
  return __builtin_amdgcn_wmma_f32_16x16x32_bf16(
      /*neg_a=*/false, a.v, /*neg_b=*/false, b.v,
      /*c_mod=*/(short)0, c, /*reuse_a=*/false, /*reuse_b=*/false);
}

// Async DMA: 16 bytes per lane, global -> LDS, tracked by ASYNCcnt.
// LDS address operand = low 32 bits of the generic shared-memory address.
__device__ __forceinline__ void async_copy16(const void* lds_dst, const void* gsrc) {
  unsigned lds_off = (unsigned)(uintptr_t)lds_dst;
  uint64_t gaddr = (uint64_t)(uintptr_t)gsrc;
  asm volatile("global_load_async_to_lds_b128 %0, %1, off"
               :: "v"(lds_off), "v"(gaddr)
               : "memory");
}

// ---------------------------------------------------------------------------
// Weight transpose + fp32 -> bf16 convert:  in[K][N] fp32  ->  out[N][K] bf16
// ---------------------------------------------------------------------------
__global__ __launch_bounds__(256)
void enc_transpose_bf16(const float* __restrict__ in, __bf16* __restrict__ out,
                        int Kd, int Nd) {
  __shared__ float tile[32][33];
  const int tx = threadIdx.x, ty = threadIdx.y;
  const int bx = blockIdx.x;   // along N
  const int by = blockIdx.y;   // along K
  for (int i = ty; i < 32; i += 8)
    tile[i][tx] = in[(size_t)(by * 32 + i) * Nd + bx * 32 + tx];
  __syncthreads();
  for (int i = ty; i < 32; i += 8)
    out[(size_t)(bx * 32 + i) * Kd + by * 32 + tx] = f2bf(tile[tx][i]);
}

// ---------------------------------------------------------------------------
// LayerNorm over rows of [M_, D_]; writes fp32 (optional) and bf16 copies.
// ---------------------------------------------------------------------------
__global__ __launch_bounds__(256)
void enc_layernorm(const float* __restrict__ x, const float* __restrict__ w,
                   const float* __restrict__ b, float* __restrict__ outF,
                   __bf16* __restrict__ outB) {
  const int row = blockIdx.x;
  const int tid = threadIdx.x;
  const int lane = tid & 31, wid = tid >> 5;
  const float* xr = x + (size_t)row * D_;
  float4 v = *(const float4*)(xr + tid * 4);
  float s = v.x + v.y + v.z + v.w;
  float q = v.x * v.x + v.y * v.y + v.z * v.z + v.w * v.w;
#pragma unroll
  for (int o = 16; o >= 1; o >>= 1) { s += __shfl_xor(s, o); q += __shfl_xor(q, o); }
  __shared__ float rs[8], rq[8], stat[2];
  if (lane == 0) { rs[wid] = s; rq[wid] = q; }
  __syncthreads();
  if (tid == 0) {
    float ts = 0.f, tq = 0.f;
    for (int i = 0; i < 8; ++i) { ts += rs[i]; tq += rq[i]; }
    float mu = ts * (1.0f / D_);
    float var = tq * (1.0f / D_) - mu * mu;
    stat[0] = mu;
    stat[1] = rsqrtf(var + 1e-5f);
  }
  __syncthreads();
  const float mu = stat[0], rstd = stat[1];
  float4 wv = *(const float4*)(w + tid * 4);
  float4 bv = *(const float4*)(b + tid * 4);
  float y0 = (v.x - mu) * rstd * wv.x + bv.x;
  float y1 = (v.y - mu) * rstd * wv.y + bv.y;
  float y2 = (v.z - mu) * rstd * wv.z + bv.z;
  float y3 = (v.w - mu) * rstd * wv.w + bv.w;
  if (outF) {
    float4 o4; o4.x = y0; o4.y = y1; o4.z = y2; o4.w = y3;
    *(float4*)(outF + (size_t)row * D_ + tid * 4) = o4;
  }
  v4bf ob;
  ob[0] = f2bf(y0); ob[1] = f2bf(y1); ob[2] = f2bf(y2); ob[3] = f2bf(y3);
  *(v4bf*)(outB + (size_t)row * D_ + tid * 4) = ob;
}

// ---------------------------------------------------------------------------
// Generic WMMA GEMM:  C[M,N] = A[M,Kd](bf16) * Bt[N,Kd](bf16, pre-transposed W)
// Block tile 128x128, K-step 32, 8 waves each computing 64x32 (4x2 WMMA tiles).
// Double-buffered LDS filled by GLOBAL_LOAD_ASYNC_TO_LDS_B128 (ASYNCcnt),
// prefetching tile k+1 while tile k runs on the WMMA pipe.
// EPI: 0 = bf16 out of (acc+bias)*alpha
//      1 = fp32 out of acc+bias+resid
//      2 = bf16 out of gelu(acc+bias)   (exact erf GELU)
// ---------------------------------------------------------------------------
template <int EPI>
__global__ __launch_bounds__(256)
void enc_gemm(const __bf16* __restrict__ A, const __bf16* __restrict__ Bt,
              const float* __restrict__ bias, const float* __restrict__ resid,
              float* __restrict__ outF, __bf16* __restrict__ outB,
              int N, int Kd, float alpha) {
  __shared__ __bf16 sA[2][128][40];   // [buf][m][k], pad keeps 16B alignment
  __shared__ __bf16 sB[2][128][40];   // [buf][n][k]
  const int tid = threadIdx.x;
  const int lane = tid & 31, wid = tid >> 5;
  const int half = lane >> 4, l16 = lane & 15;
  const int wm0 = (wid >> 2) * 64;
  const int wn0 = (wid & 3) * 32;
  const int row0 = blockIdx.y * 128, col0 = blockIdx.x * 128;

  // Per thread: 2 A-chunks + 2 B-chunks of 16B per tile => 4 async ops.
  auto issue_tile = [&](int buf, int k0) {
#pragma unroll
    for (int j = 0; j < 2; ++j) {
      int c = tid + 256 * j;          // 512 chunks of 8 bf16 per tile
      int r = c >> 2, cc = (c & 3) * 8;
      async_copy16(&sA[buf][r][cc], &A[(size_t)(row0 + r) * Kd + k0 + cc]);
      async_copy16(&sB[buf][r][cc], &Bt[(size_t)(col0 + r) * Kd + k0 + cc]);
    }
  };

  v8f acc[4][2];
#pragma unroll
  for (int i = 0; i < 4; ++i)
#pragma unroll
    for (int j = 0; j < 2; ++j) acc[i][j] = {};

  issue_tile(0, 0);   // prologue prefetch

  for (int k0 = 0; k0 < Kd; k0 += 32) {
    const int cur = (k0 >> 5) & 1;
    // All waves consumed buffer `cur^1` (their ds_load results were waited on
    // before the WMMAs of the previous iteration) -> safe to overwrite it.
    __syncthreads();
    if (k0 + 32 < Kd) {
      issue_tile(cur ^ 1, k0 + 32);
      // in-order ASYNCcnt: <=4 outstanding means current tile has landed
      asm volatile("s_wait_asynccnt 0x4" ::: "memory");
    } else {
      asm volatile("s_wait_asynccnt 0x0" ::: "memory");
    }
    __syncthreads();   // every wave's chunks of `cur` are in LDS

    Frag fa[4], fb[2];
#pragma unroll
    for (int mt = 0; mt < 4; ++mt) {
      int r = wm0 + mt * 16 + l16;
      fa[mt].h[0] = *(const v8bf*)&sA[cur][r][half * 8];        // K 0-7 / 8-15
      fa[mt].h[1] = *(const v8bf*)&sA[cur][r][16 + half * 8];   // K 16-23 / 24-31
    }
#pragma unroll
    for (int nt = 0; nt < 2; ++nt) {
      int cidx = wn0 + nt * 16 + l16;
      fb[nt].h[0] = *(const v8bf*)&sB[cur][cidx][half * 8];       // K 0-7 / 8-15
      fb[nt].h[1] = *(const v8bf*)&sB[cur][cidx][16 + half * 8];  // K 16-23 / 24-31
    }
#pragma unroll
    for (int mt = 0; mt < 4; ++mt)
#pragma unroll
      for (int nt = 0; nt < 2; ++nt)
        acc[mt][nt] = wmma_bf16(fa[mt], fb[nt], acc[mt][nt]);
  }

#pragma unroll
  for (int nt = 0; nt < 2; ++nt) {
    int col = col0 + wn0 + nt * 16 + l16;
    float bcol = bias[col];
#pragma unroll
    for (int mt = 0; mt < 4; ++mt) {
#pragma unroll
      for (int i = 0; i < 8; ++i) {
        int row = row0 + wm0 + mt * 16 + half * 8 + i;  // C layout: m = 8*half+i
        size_t idx = (size_t)row * N + col;
        float t = (acc[mt][nt][i] + bcol) * alpha;
        if (EPI == 0) {
          outB[idx] = f2bf(t);
        } else if (EPI == 1) {
          outF[idx] = t + resid[idx];
        } else {
          float g = 0.5f * t * (1.0f + erff(t * 0.70710678118654752f));
          outB[idx] = f2bf(g);
        }
      }
    }
  }
}

// ---------------------------------------------------------------------------
// Fused flash attention.  Q/K/V live in [B*S, D] bf16 buffers; per (b,h) the
// matrix is [S, DK] with row-stride D, column offset h*DK.  Q is pre-scaled.
// Block = 64 q rows (4 waves x 16), loop over 64-wide K chunks.
// K tile is DMA'd with async-to-LDS; V needs an elementwise transpose so it
// goes through VGPR + ds_store.
// ---------------------------------------------------------------------------
__global__ __launch_bounds__(128)
void enc_attn(const __bf16* __restrict__ qb, const __bf16* __restrict__ kb,
              const __bf16* __restrict__ vb, const float* __restrict__ bias,
              __bf16* __restrict__ ctx) {
  __shared__ __bf16 sK[64][72];       // [k_row][d]   == Bt layout for Q*K^T
  __shared__ __bf16 sV[64][72];       // [d][k_row]   == Bt layout for P*V
  __shared__ __bf16 sP[4][16][72];    // per-wave P transpose buffer
  const int tid = threadIdx.x;
  const int lane = tid & 31, wid = tid >> 5;
  const int half = lane >> 4, l16 = lane & 15;
  const int b = blockIdx.y / H_, h = blockIdx.y % H_;
  const size_t base = ((size_t)b * S_) * D_ + (size_t)h * DK_;
  const int myq0 = blockIdx.x * 64 + wid * 16;
  const float* bias_bh = bias + ((size_t)blockIdx.y * S_) * S_;

  // Q fragments (A layout, DK=64 -> 2 frags), held for the whole loop
  Frag fq[2];
#pragma unroll
  for (int g = 0; g < 2; ++g) {
    const __bf16* qp = qb + base + (size_t)(myq0 + l16) * D_ + g * 32;
    fq[g].h[0] = *(const v8bf*)(qp + half * 8);
    fq[g].h[1] = *(const v8bf*)(qp + 16 + half * 8);
  }

  float rmax[8], rsum[8];
  v8f o[4];
#pragma unroll
  for (int i = 0; i < 8; ++i) { rmax[i] = -1e30f; rsum[i] = 0.f; }
#pragma unroll
  for (int nt = 0; nt < 4; ++nt) o[nt] = {};

  for (int kc0 = 0; kc0 < S_; kc0 += 64) {
    __syncthreads();
#pragma unroll
    for (int j = 0; j < 4; ++j) {       // 512 chunks of 8 bf16, 128 threads
      int c = tid + 128 * j;
      int r = c >> 3, cc = (c & 7) * 8;
      async_copy16(&sK[r][cc], &kb[base + (size_t)(kc0 + r) * D_ + cc]);
      v8bf vv = *(const v8bf*)&vb[base + (size_t)(kc0 + r) * D_ + cc];
#pragma unroll
      for (int e = 0; e < 8; ++e) sV[cc + e][r] = vv[e];   // transpose V
    }
    asm volatile("s_wait_asynccnt 0x0" ::: "memory");
    __syncthreads();

    // scores S[16 x 64] = Q * K^T + bias
    v8f s[4];
#pragma unroll
    for (int nt = 0; nt < 4; ++nt) {
      int col = nt * 16 + l16;
      v8f c = {};
#pragma unroll
      for (int g = 0; g < 2; ++g) {
        Frag fb_;
        fb_.h[0] = *(const v8bf*)&sK[col][g * 32 + half * 8];
        fb_.h[1] = *(const v8bf*)&sK[col][g * 32 + 16 + half * 8];
        c = wmma_bf16(fq[g], fb_, c);
      }
      const float* bp = bias_bh + (size_t)(myq0 + half * 8) * S_ + kc0 + col;
#pragma unroll
      for (int i = 0; i < 8; ++i) c[i] += bp[(size_t)i * S_];
      s[nt] = c;
    }

    // online softmax (row m = 8*half + i lives across the 16 lanes of a half)
    float scl[8];
#pragma unroll
    for (int i = 0; i < 8; ++i) {
      float m = fmaxf(fmaxf(s[0][i], s[1][i]), fmaxf(s[2][i], s[3][i]));
#pragma unroll
      for (int off = 8; off >= 1; off >>= 1) m = fmaxf(m, __shfl_xor(m, off));
      float nm = fmaxf(rmax[i], m);
      scl[i] = __expf(rmax[i] - nm);
      rmax[i] = nm;
      float a = 0.f;
#pragma unroll
      for (int nt = 0; nt < 4; ++nt) {
        float p = __expf(s[nt][i] - nm);
        s[nt][i] = p;
        a += p;
      }
#pragma unroll
      for (int off = 8; off >= 1; off >>= 1) a += __shfl_xor(a, off);
      rsum[i] = rsum[i] * scl[i] + a;
    }
#pragma unroll
    for (int nt = 0; nt < 4; ++nt)
#pragma unroll
      for (int i = 0; i < 8; ++i) o[nt][i] *= scl[i];

    // P: C layout -> bf16 -> per-wave LDS -> reload in A layout
#pragma unroll
    for (int nt = 0; nt < 4; ++nt)
#pragma unroll
      for (int i = 0; i < 8; ++i)
        sP[wid][half * 8 + i][nt * 16 + l16] = f2bf(s[nt][i]);
#if __has_builtin(__builtin_amdgcn_wave_barrier)
    __builtin_amdgcn_wave_barrier();   // keep DS write->read order (in-order per wave)
#endif
#pragma unroll
    for (int g = 0; g < 2; ++g) {
      Frag fp;
      fp.h[0] = *(const v8bf*)&sP[wid][l16][g * 32 + half * 8];
      fp.h[1] = *(const v8bf*)&sP[wid][l16][g * 32 + 16 + half * 8];
#pragma unroll
      for (int nt = 0; nt < 4; ++nt) {
        Frag fv;
        int dcol = nt * 16 + l16;
        fv.h[0] = *(const v8bf*)&sV[dcol][g * 32 + half * 8];
        fv.h[1] = *(const v8bf*)&sV[dcol][g * 32 + 16 + half * 8];
        o[nt] = wmma_bf16(fp, fv, o[nt]);
      }
    }
  }

#pragma unroll
  for (int i = 0; i < 8; ++i) {
    float inv = 1.0f / rsum[i];
#pragma unroll
    for (int nt = 0; nt < 4; ++nt)
      ctx[base + (size_t)(myq0 + half * 8 + i) * D_ + nt * 16 + l16] =
          f2bf(o[nt][i] * inv);
  }
}

// ---------------------------------------------------------------------------
extern "C" void kernel_launch(void* const* d_in, const int* in_sizes, int n_in,
                              void* d_out, int out_size, void* d_ws, size_t ws_size,
                              hipStream_t stream) {
  (void)in_sizes; (void)n_in; (void)out_size; (void)ws_size;
  const float* x         = (const float*)d_in[0];
  const float* attn_bias = (const float*)d_in[1];
  const float* ln1_w = (const float*)d_in[2];
  const float* ln1_b = (const float*)d_in[3];
  const float* wq = (const float*)d_in[4];
  const float* bq = (const float*)d_in[5];
  const float* wk = (const float*)d_in[6];
  const float* bk = (const float*)d_in[7];
  const float* wv = (const float*)d_in[8];
  const float* bv = (const float*)d_in[9];
  const float* wo = (const float*)d_in[10];
  const float* bo = (const float*)d_in[11];
  const float* ln2_w = (const float*)d_in[12];
  const float* ln2_b = (const float*)d_in[13];
  const float* w1 = (const float*)d_in[14];
  const float* b1 = (const float*)d_in[15];
  const float* w2 = (const float*)d_in[16];
  const float* b2 = (const float*)d_in[17];
  float* out = (float*)d_out;

  char* wsb = (char*)d_ws;
  size_t off = 0;
  auto alloc = [&](size_t bytes) -> char* {
    char* p = wsb + off;
    off += (bytes + 255) & ~(size_t)255;
    return p;
  };
  __bf16* wqT  = (__bf16*)alloc((size_t)D_ * D_ * 2);
  __bf16* wkT  = (__bf16*)alloc((size_t)D_ * D_ * 2);
  __bf16* wvT  = (__bf16*)alloc((size_t)D_ * D_ * 2);
  __bf16* woT  = (__bf16*)alloc((size_t)D_ * D_ * 2);
  __bf16* w1T  = (__bf16*)alloc((size_t)D_ * F_ * 2);
  __bf16* w2T  = (__bf16*)alloc((size_t)F_ * D_ * 2);
  float*  normF = (float*)alloc((size_t)M_ * D_ * 4);
  __bf16* normB = (__bf16*)alloc((size_t)M_ * D_ * 2);
  __bf16* qbf  = (__bf16*)alloc((size_t)M_ * D_ * 2);
  __bf16* kbf  = (__bf16*)alloc((size_t)M_ * D_ * 2);
  __bf16* vbf  = (__bf16*)alloc((size_t)M_ * D_ * 2);
  __bf16* ctxb = (__bf16*)alloc((size_t)M_ * D_ * 2);
  float*  z1   = (float*)alloc((size_t)M_ * D_ * 4);
  __bf16* hbf  = (__bf16*)alloc((size_t)M_ * D_ * 2);
  __bf16* midb = (__bf16*)alloc((size_t)M_ * F_ * 2);

  const dim3 tb(32, 8);
  // weights -> bf16, transposed to [N][K]
  enc_transpose_bf16<<<dim3(D_ / 32, D_ / 32), tb, 0, stream>>>(wq, wqT, D_, D_);
  enc_transpose_bf16<<<dim3(D_ / 32, D_ / 32), tb, 0, stream>>>(wk, wkT, D_, D_);
  enc_transpose_bf16<<<dim3(D_ / 32, D_ / 32), tb, 0, stream>>>(wv, wvT, D_, D_);
  enc_transpose_bf16<<<dim3(D_ / 32, D_ / 32), tb, 0, stream>>>(wo, woT, D_, D_);
  enc_transpose_bf16<<<dim3(F_ / 32, D_ / 32), tb, 0, stream>>>(w1, w1T, D_, F_);
  enc_transpose_bf16<<<dim3(D_ / 32, F_ / 32), tb, 0, stream>>>(w2, w2T, F_, D_);

  // LN1
  enc_layernorm<<<M_, 256, 0, stream>>>(x, ln1_w, ln1_b, normF, normB);

  // Q/K/V projections (Q pre-scaled by DK^-0.5)
  enc_gemm<0><<<dim3(D_ / 128, M_ / 128), 256, 0, stream>>>(
      normB, wqT, bq, nullptr, nullptr, qbf, D_, D_, 0.125f);
  enc_gemm<0><<<dim3(D_ / 128, M_ / 128), 256, 0, stream>>>(
      normB, wkT, bk, nullptr, nullptr, kbf, D_, D_, 1.0f);
  enc_gemm<0><<<dim3(D_ / 128, M_ / 128), 256, 0, stream>>>(
      normB, wvT, bv, nullptr, nullptr, vbf, D_, D_, 1.0f);

  // fused attention
  enc_attn<<<dim3(S_ / 64, B_ * H_), 128, 0, stream>>>(qbf, kbf, vbf, attn_bias, ctxb);

  // output projection + residual (z1 = norm_x + attn_out)
  enc_gemm<1><<<dim3(D_ / 128, M_ / 128), 256, 0, stream>>>(
      ctxb, woT, bo, normF, z1, nullptr, D_, D_, 1.0f);

  // LN2
  enc_layernorm<<<M_, 256, 0, stream>>>(z1, ln2_w, ln2_b, nullptr, hbf);

  // FFN
  enc_gemm<2><<<dim3(F_ / 128, M_ / 128), 256, 0, stream>>>(
      hbf, w1T, b1, nullptr, nullptr, midb, F_, D_, 1.0f);
  enc_gemm<1><<<dim3(D_ / 128, M_ / 128), 256, 0, stream>>>(
      midb, w2T, b2, z1, out, nullptr, D_, F_, 1.0f);
}
